// ConditionalFNP_69088843924154
// MI455X (gfx1250) — compile-verified, matrix operand order checked by reading, not run on hardware
//
#include <hip/hip_runtime.h>
#include <cstdint>

typedef __attribute__((ext_vector_type(16))) _Float16 v16h;
typedef __attribute__((ext_vector_type(8)))  float    v8f;
typedef _Float16 half_t;

#define B_SZ   64
#define S_SZ   4096
#define E_DIM  266
#define E_PAD  288
#define HIDD   512
#define NH     128
#define M_ROWS (B_SZ * NH)   /* 8192 */

#define BM 128
#define BN 64
#define KT 32
#define LDA 36   /* 32 + 4 halves pad: spreads LDS banks */
#define LDB 36

union Frag { v16h v; uint32_t u[8]; };

// ---------------- prep: A0 = f16( z[b] + emb[h] ), K-padded to 288 ----------------
__global__ void fnp_prep_a0(const float* __restrict__ z, const float* __restrict__ emb,
                            half_t* __restrict__ A0) {
  const int row = blockIdx.x;           // 0..8191
  const int b = row >> 7, h = row & 127;
  for (int e = threadIdx.x; e < E_PAD; e += blockDim.x) {
    float v = (e < E_DIM) ? (z[b * E_DIM + e] + emb[h * E_DIM + e]) : 0.0f;
    A0[(size_t)row * E_PAD + e] = (half_t)v;
  }
}

// ---------------- prep: Wt_in[d][e] = w_in[e][d] (f16, K-padded) ----------------
__global__ void fnp_prep_win(const float* __restrict__ w_in, half_t* __restrict__ Wt) {
  const int idx = blockIdx.x * blockDim.x + threadIdx.x;
  if (idx >= HIDD * E_PAD) return;
  const int d = idx / E_PAD, e = idx % E_PAD;
  Wt[idx] = (half_t)((e < E_DIM) ? w_in[(size_t)e * HIDD + d] : 0.0f);
}

// ---------------- prep: WtH[i][n][k] = w_h[i][k][n] (f16) ----------------
__global__ void fnp_prep_wh(const float* __restrict__ w_h, half_t* __restrict__ WtH) {
  const int idx = blockIdx.x * blockDim.x + threadIdx.x;
  if (idx >= 3 * HIDD * HIDD) return;
  const int i = idx >> 18;
  const int r = idx & ((1 << 18) - 1);
  const int n = r >> 9, k = r & 511;
  WtH[idx] = (half_t)w_h[((size_t)(i * HIDD + k)) * HIDD + n];
}

// ---------------- WMMA GEMM + bias + SiLU: C(MxN,f16) = silu(A(MxK) * Bt^T + bias) ----------------
// A: row-major [M][Kpad] f16.  Bt: row-major [N][Kpad] f16 (i.e. B transposed).
__global__ void __launch_bounds__(256)
fnp_gemm_silu(const half_t* __restrict__ A, const half_t* __restrict__ Bt,
              const float* __restrict__ bias, half_t* __restrict__ C,
              int N, int Kpad, int Ksteps) {
  __shared__ half_t lA[BM * LDA];
  __shared__ half_t lB[BN * LDB];

  const int tid  = threadIdx.x;
  const int lane = tid & 31;
  const int wid  = tid >> 5;                 // 8 waves
  const int wm   = (wid >> 1) * 32;          // wave M origin in tile: 0,32,64,96
  const int wn   = (wid & 1) * 32;           // wave N origin in tile: 0,32
  const int mBase = blockIdx.y * BM;
  const int nBase = blockIdx.x * BN;

  const int lm = lane & 15;                  // matrix row/col within 16-lane group
  const int kh = lane >> 4;                  // K-half select per ISA layout

  v8f acc[2][2] = {};

  for (int ks = 0; ks < Ksteps; ++ks) {
    const int k0 = ks * KT;

    // ---- stage A tile: 128 rows x 32 halves (512 x 16B chunks, 2 per thread) ----
    #pragma unroll
    for (int qq = 0; qq < 2; ++qq) {
      const int q  = tid + qq * 256;         // 0..511
      const int r  = q >> 2;                 // 0..127
      const int ch = (q & 3) * 8;            // 0,8,16,24 halves
      const uint4 v = *(const uint4*)(A + (size_t)(mBase + r) * Kpad + k0 + ch);
      *(uint2*)&lA[r * LDA + ch]     = make_uint2(v.x, v.y);
      *(uint2*)&lA[r * LDA + ch + 4] = make_uint2(v.z, v.w);
    }
    // ---- stage B tile: 64 rows(n) x 32 halves(k) (256 x 16B chunks) ----
    {
      const int r  = tid >> 2;               // 0..63
      const int ch = (tid & 3) * 8;
      const uint4 v = *(const uint4*)(Bt + (size_t)(nBase + r) * Kpad + k0 + ch);
      *(uint2*)&lB[r * LDB + ch]     = make_uint2(v.x, v.y);
      *(uint2*)&lB[r * LDB + ch + 4] = make_uint2(v.z, v.w);
    }
    __syncthreads();

    // ---- assemble fragments per ISA 16-bit layout:
    // VGPR pair j holds K = 2*(j%4) + 8*kh + 16*(j/4), elements (k,k+1)
    Frag a[2], b[2];
    #pragma unroll
    for (int j = 0; j < 8; ++j) {
      const int k = 2 * (j & 3) + 8 * kh + 16 * (j >> 2);
      a[0].u[j] = *(const uint32_t*)&lA[(wm +      lm) * LDA + k];
      a[1].u[j] = *(const uint32_t*)&lA[(wm + 16 + lm) * LDA + k];
      b[0].u[j] = *(const uint32_t*)&lB[(wn +      lm) * LDB + k];
      b[1].u[j] = *(const uint32_t*)&lB[(wn + 16 + lm) * LDB + k];
    }

    #pragma unroll
    for (int i = 0; i < 2; ++i)
      #pragma unroll
      for (int j = 0; j < 2; ++j)
        acc[i][j] = __builtin_amdgcn_wmma_f32_16x16x32_f16(
            false, a[i].v, false, b[j].v, (short)0, acc[i][j], false, false);

    __syncthreads();
  }

  // ---- epilogue: bias + SiLU, store f16. C layout: VGPR vv -> M = vv + 8*kh, lane%16 -> N
  #pragma unroll
  for (int i = 0; i < 2; ++i) {
    #pragma unroll
    for (int j = 0; j < 2; ++j) {
      const int col = nBase + wn + j * 16 + lm;
      const float bv = bias[col];
      #pragma unroll
      for (int vv = 0; vv < 8; ++vv) {
        const int row = mBase + wm + i * 16 + vv + 8 * kh;
        float x = acc[i][j][vv] + bv;
        x = x * (1.0f / (1.0f + __expf(-x)));     // SiLU
        C[(size_t)row * N + col] = (half_t)x;
      }
    }
  }
}

// ---------------- head: cf[row] = (h . w_out[:,0] + b0, h . w_out[:,1] + b1) ----------------
__global__ void fnp_coeffs(const half_t* __restrict__ H, const float* __restrict__ w_out,
                           const float* __restrict__ b_out, float2* __restrict__ cf) {
  const int wid = threadIdx.x >> 5, lane = threadIdx.x & 31;
  const int row = blockIdx.x * 8 + wid;
  float s0 = 0.0f, s1 = 0.0f;
  for (int d = lane; d < HIDD; d += 32) {
    const float h = (float)H[(size_t)row * HIDD + d];
    s0 = fmaf(h, w_out[d * 2 + 0], s0);
    s1 = fmaf(h, w_out[d * 2 + 1], s1);
  }
  #pragma unroll
  for (int off = 16; off > 0; off >>= 1) {
    s0 += __shfl_down(s0, off, 32);
    s1 += __shfl_down(s1, off, 32);
  }
  if (lane == 0) cf[row] = make_float2(s0 + b_out[0], s1 + b_out[1]);
}

// ---------------- synthesis: out[b,s] = sum_h cos(h*th)*cos_c + sin(h*th)*sin_c ----------------
// FREQS = 1..128 are integer harmonics -> angle-addition recurrence, one sincos per thread.
__global__ void fnp_synth(const float* __restrict__ tx, const float2* __restrict__ cf,
                          float* __restrict__ out) {
  __shared__ float2 sc[NH];
  const int b = blockIdx.y;
  if (threadIdx.x < NH) sc[threadIdx.x] = cf[b * NH + threadIdx.x];
  __syncthreads();
  const int s = blockIdx.x * blockDim.x + threadIdx.x;
  const float theta = 6.28318530717958647692f * tx[(size_t)b * S_SZ + s];
  float s1, c1;
  __sincosf(theta, &s1, &c1);
  float cs = c1, sn = s1, accv = 0.0f;
  #pragma unroll 8
  for (int h = 0; h < NH; ++h) {
    accv = fmaf(cs, sc[h].y, fmaf(sn, sc[h].x, accv));   // cos_c = .y, sin_c = .x
    const float nc = fmaf(cs, c1, -(sn * s1));
    const float ns = fmaf(sn, c1, cs * s1);
    cs = nc; sn = ns;
  }
  out[(size_t)b * S_SZ + s] = accv;
}

extern "C" void kernel_launch(void* const* d_in, const int* in_sizes, int n_in,
                              void* d_out, int out_size, void* d_ws, size_t ws_size,
                              hipStream_t stream) {
  const float* target_x = (const float*)d_in[0];
  const float* z        = (const float*)d_in[1];
  /* d_in[2] = x : unused by reference */
  const float* emb      = (const float*)d_in[3];
  const float* w_in     = (const float*)d_in[4];
  const float* b_in     = (const float*)d_in[5];
  const float* w_h      = (const float*)d_in[6];
  const float* b_h      = (const float*)d_in[7];
  const float* w_out    = (const float*)d_in[8];
  const float* b_out    = (const float*)d_in[9];
  float* out = (float*)d_out;

  char* ws = (char*)d_ws;
  size_t off = 0;
  auto alloc = [&](size_t bytes) {
    char* p = ws + off;
    off = (off + bytes + 255) & ~(size_t)255;
    return p;
  };
  half_t* A0   = (half_t*)alloc((size_t)M_ROWS * E_PAD * sizeof(half_t));  // 4.5 MB
  half_t* WtIn = (half_t*)alloc((size_t)HIDD * E_PAD * sizeof(half_t));    // 288 KB
  half_t* WtH  = (half_t*)alloc((size_t)3 * HIDD * HIDD * sizeof(half_t)); // 1.5 MB
  half_t* H1   = (half_t*)alloc((size_t)M_ROWS * HIDD * sizeof(half_t));   // 8 MB
  half_t* H2   = (half_t*)alloc((size_t)M_ROWS * HIDD * sizeof(half_t));   // 8 MB
  float2* CF   = (float2*)alloc((size_t)M_ROWS * sizeof(float2));          // 64 KB

  fnp_prep_a0 <<<M_ROWS, 128, 0, stream>>>(z, emb, A0);
  fnp_prep_win<<<(HIDD * E_PAD + 255) / 256, 256, 0, stream>>>(w_in, WtIn);
  fnp_prep_wh <<<(3 * HIDD * HIDD + 255) / 256, 256, 0, stream>>>(w_h, WtH);

  dim3 g(HIDD / BN, M_ROWS / BM);   // (8, 64) blocks of 256 threads
  fnp_gemm_silu<<<g, 256, 0, stream>>>(A0, WtIn,               b_in,         H1, HIDD, E_PAD, E_PAD / KT);
  fnp_gemm_silu<<<g, 256, 0, stream>>>(H1, WtH,                b_h,          H2, HIDD, HIDD,  HIDD / KT);
  fnp_gemm_silu<<<g, 256, 0, stream>>>(H2, WtH + HIDD * HIDD,  b_h + HIDD,   H1, HIDD, HIDD,  HIDD / KT);
  fnp_gemm_silu<<<g, 256, 0, stream>>>(H1, WtH + 2*HIDD*HIDD,  b_h + 2*HIDD, H2, HIDD, HIDD,  HIDD / KT);

  fnp_coeffs<<<M_ROWS / 8, 256, 0, stream>>>(H2, w_out, b_out, CF);
  fnp_synth <<<dim3(S_SZ / 256, B_SZ), 256, 0, stream>>>(target_x, CF, out);
}
